// MobiusFlow_65103114273161
// MI455X (gfx1250) — matrix operand order repaired and checked by reading, not compiled, not run on hardware
//
#include <hip/hip_runtime.h>
#include <hip/hip_bf16.h>
#include <math.h>

// ---------------------------------------------------------------------------
// MobiusFlow fused kernel for gfx1250 (MI455X).
//  - 2-layer MLP done with v_wmma_f32_16x16x32_f16 (f16 in, f32 accum)
//  - per-block tile: 16 samples, 256 threads (8 wave32)
//  - weights pre-swizzled into WMMA B-fragment layout by a prep kernel
// ---------------------------------------------------------------------------

typedef __attribute__((ext_vector_type(16))) _Float16 v16h;
typedef __attribute__((ext_vector_type(8)))  _Float16 v8h;
typedef __attribute__((ext_vector_type(8)))  float    v8f;

#define TWO_PI_F 6.283185307179586f

// IN_DIM = 262 padded to 288 = 9*32 ; HID = 256 = 8*32 ; OUT_DIM = 262 padded to 272 = 17*16
#define KK1 9              // k-steps GEMM1
#define KK2 8              // k-steps GEMM2
#define NT1 16             // n-tiles GEMM1 (256/16)
#define NT2 17             // n-tiles GEMM2 (272/16)
#define W1F_ELEMS (NT1*KK1*32*16)   // 73728 halves
#define W2F_ELEMS (NT2*KK2*32*16)   // 69632 halves

// -------------------------------------------------------------------------
// Pre-swizzle W1 (262x256) and W2 (256x262) into per-lane B fragments.
// Fragment layout (contiguous k order per lane half, same as the A loads):
//   lane l<16 : column n = tile*16+l, halves h=0..15 <-> K = kk*32 + h
//   lane l>=16: column n = tile*16+(l-16), K = kk*32 + 16 + h
// Stored so each lane reads 16 contiguous halves (32 B) per (tile,kk).
// -------------------------------------------------------------------------
__global__ void __launch_bounds__(256)
prep_weights(const float* __restrict__ W1, const float* __restrict__ W2,
             _Float16* __restrict__ w1f, _Float16* __restrict__ w2f)
{
    int idx = blockIdx.x * 256 + threadIdx.x;
    if (idx < W1F_ELEMS) {
        int h    = idx & 15;
        int lane = (idx >> 4) & 31;
        int tk   = idx >> 9;              // t*9 + kk
        int kk   = tk % KK1;
        int t    = tk / KK1;
        int n    = t * 16 + (lane & 15);
        int k    = kk * 32 + (lane >> 4) * 16 + h;
        w1f[idx] = (k < 262) ? (_Float16)W1[k * 256 + n] : (_Float16)0.0f;
    }
    int idx2 = idx - W1F_ELEMS;
    if (idx2 >= 0 && idx2 < W2F_ELEMS) {
        int h    = idx2 & 15;
        int lane = (idx2 >> 4) & 31;
        int tk   = idx2 >> 9;             // t*8 + kk
        int kk   = tk & 7;
        int t    = tk >> 3;
        int n    = t * 16 + (lane & 15);
        int k    = kk * 32 + (lane >> 4) * 16 + h;   // k < 256 always
        w2f[idx2] = (n < 262) ? (_Float16)W2[k * 262 + n] : (_Float16)0.0f;
    }
}

// -------------------------------------------------------------------------
// Fused MLP + Mobius flow.
// -------------------------------------------------------------------------
__global__ void __launch_bounds__(256)
mobius_fused(const float* __restrict__ rot, const float* __restrict__ vec,
             const float* __restrict__ feat, const int* __restrict__ perm,
             const float* __restrict__ b1, const float* __restrict__ b2,
             const _Float16* __restrict__ w1f, const _Float16* __restrict__ w2f,
             float* __restrict__ out, int N)
{
    // Row strides padded so A-fragment b128 reads from the 16 rows hit
    // distinct bank groups (592 B -> 148 dw, 528 B -> 132 dw strides).
    __shared__ __align__(16) _Float16 sIn[16][296];    // cond_in tile (f16)
    __shared__ __align__(16) _Float16 sHid[16][264];   // hidden tile  (f16)
    __shared__ __align__(16) float    sConds[16][272]; // conds tile   (f32)

    const int tid = threadIdx.x;
    const int n0  = blockIdx.x * 16;
    const int p0 = perm[0], p1 = perm[1], p2 = perm[2];

    // ---- stage cond_in = [y(3) | feature(256) | vector[:3]] padded to 288
    for (int idx = tid; idx < 16 * 288; idx += 256) {
        int row = idx / 288, col = idx % 288;
        int n = n0 + row;
        float val;
        if      (col < 3)   val = rot[n * 9 + col * 3 + p1];
        else if (col < 259) val = feat[n * 256 + (col - 3)];
        else if (col < 262) val = vec[n * 6 + (col - 259)];
        else                val = 0.0f;
        sIn[row][col] = (_Float16)val;
    }
    __syncthreads();

    const int w    = tid >> 5;        // wave id 0..7 (uniform per wave)
    const int lane = tid & 31;
    const int g    = lane >> 4;       // lane group for A fragment
    const int m    = lane & 15;       // row for A fragment
    const int mb   = g * 8;           // C fragment row base
    const int nn   = m;               // C fragment column-in-tile

    // ---- GEMM1: hid(16x256) = cond_in(16x288) @ W1p(288x256), 2 n-tiles/wave
    v8f acc0 = {}, acc1 = {};
    const int t0 = 2 * w, t1 = 2 * w + 1;
    for (int kk = 0; kk < KK1; ++kk) {
        union { v16h v; v8h h[2]; } a;
        int base = kk * 32 + g * 16;
        a.h[0] = *(const v8h*)&sIn[m][base];
        a.h[1] = *(const v8h*)&sIn[m][base + 8];
        v16h bA = *(const v16h*)(w1f + ((size_t)(t0 * KK1 + kk) * 32 + lane) * 16);
        v16h bB = *(const v16h*)(w1f + ((size_t)(t1 * KK1 + kk) * 32 + lane) * 16);
        acc0 = __builtin_amdgcn_wmma_f32_16x16x32_f16(false, a.v, false, bA,
                                                      (short)0, acc0, false, false);
        acc1 = __builtin_amdgcn_wmma_f32_16x16x32_f16(false, a.v, false, bB,
                                                      (short)0, acc1, false, false);
    }
    #pragma unroll
    for (int j = 0; j < 8; ++j) {                       // bias + relu -> LDS f16
        int na = t0 * 16 + nn;
        sHid[mb + j][na] = (_Float16)fmaxf(acc0[j] + b1[na], 0.0f);
        int nb = t1 * 16 + nn;
        sHid[mb + j][nb] = (_Float16)fmaxf(acc1[j] + b1[nb], 0.0f);
    }
    __syncthreads();

    // ---- GEMM2: conds(16x272) = hid(16x256) @ W2p(256x272), 17 n-tiles
    for (int t = w; t < NT2; t += 8) {
        v8f acc = {};
        for (int kk = 0; kk < KK2; ++kk) {
            union { v16h v; v8h h[2]; } a;
            int base = kk * 32 + g * 16;
            a.h[0] = *(const v8h*)&sHid[m][base];
            a.h[1] = *(const v8h*)&sHid[m][base + 8];
            v16h bB = *(const v16h*)(w2f + ((size_t)(t * KK2 + kk) * 32 + lane) * 16);
            acc = __builtin_amdgcn_wmma_f32_16x16x32_f16(false, a.v, false, bB,
                                                         (short)0, acc, false, false);
        }
        #pragma unroll
        for (int j = 0; j < 8; ++j) {
            int nc = t * 16 + nn;
            float cv = acc[j];
            if (nc < 262) cv += b2[nc];
            sConds[mb + j][nc] = cv;
        }
    }
    __syncthreads();

    // ---- Mobius epilogue: 16 lanes per sample, 4 of K=64 terms each
    const int s  = tid >> 4;     // sample in tile
    const int jj = tid & 15;     // worker within sample
    const int n  = n0 + s;

    float x0 = rot[n*9 + 0 + p0], x1 = rot[n*9 + 3 + p0], x2 = rot[n*9 + 6 + p0];
    float y0 = rot[n*9 + 0 + p1], y1 = rot[n*9 + 3 + p1], y2 = rot[n*9 + 6 + p1];
    float ixn = rsqrtf(x0*x0 + x1*x1 + x2*x2);
    float r0 = -x0*ixn, r1 = -x1*ixn, r2 = -x2*ixn;
    float v0 = y1*r2 - y2*r1, v1 = y2*r0 - y0*r2, v2 = y0*r1 - y1*r0;
    float ivn = rsqrtf(v0*v0 + v1*v1 + v2*v2);
    v0 *= ivn; v1 *= ivn; v2 *= ivn;
    float theta = atan2f(x0*v0 + x1*v1 + x2*v2, x0*r0 + x1*r1 + x2*r2);
    float st = sinf(theta), ct = cosf(theta);
    float dz0 = -st*r0 + ct*v0, dz1 = -st*r1 + ct*v1, dz2 = -st*r2 + ct*v2;

    float acc_t = 0.f, acc_d = 0.f, acc_w = 0.f;
    #pragma unroll
    for (int kq = 0; kq < 4; ++kq) {
        int k = jj + kq * 16;
        float c  = sConds[s][k];
        float sw = fmaxf(c, 0.f) + log1pf(expf(-fabsf(c)));   // softplus
        float wa = sConds[s][64 + k*3 + 0];
        float wb = sConds[s][64 + k*3 + 1];
        float wc = sConds[s][64 + k*3 + 2];
        float dyw = y0*wa + y1*wb + y2*wc;                    // (I - yy^T) w
        wa -= dyw*y0; wb -= dyw*y1; wc -= dyw*y2;
        float wnrm = sqrtf(wa*wa + wb*wb + wc*wc);
        float sc = 0.7f / (1.0f + wnrm);
        wa *= sc; wb *= sc; wc *= sc;
        float wn2 = wa*wa + wb*wb + wc*wc;
        float za = x0 - wa, zb = x1 - wb, zc = x2 - wc;
        float z2 = za*za + zb*zb + zc*zc;
        float f  = (1.0f - wn2) / z2;
        float h0 = f*za - wa, h1 = f*zb - wb, h2 = f*zc - wc;
        float rad = atan2f(h0*v0 + h1*v1 + h2*v2, h0*r0 + h1*r1 + h2*r2);
        rad += (rad < 0.f) ? TWO_PI_F : 0.f;
        float izn = rsqrtf(z2);
        float u0 = za*izn, u1 = zb*izn, u2 = zc*izn;
        float udz = u0*dz0 + u1*dz1 + u2*dz2;
        float dh0 = f*(dz0 - 2.f*udz*u0);
        float dh1 = f*(dz1 - 2.f*udz*u1);
        float dh2 = f*(dz2 - 2.f*udz*u2);
        float dtk = sqrtf(dh0*dh0 + dh1*dh1 + dh2*dh2);
        acc_t += sw * rad;
        acc_d += sw * dtk;
        acc_w += sw;
    }
    #pragma unroll
    for (int off = 8; off > 0; off >>= 1) {                  // 16-lane reduce
        acc_t += __shfl_xor(acc_t, off, 16);
        acc_d += __shfl_xor(acc_d, off, 16);
        acc_w += __shfl_xor(acc_w, off, 16);
    }
    float tt  = acc_t / acc_w;
    float ldj = logf(acc_d / acc_w);
    float cte = cosf(tt), ste = sinf(tt);
    float tx0 = r0*cte + v0*ste, tx1 = r1*cte + v1*ste, tx2 = r2*cte + v2*ste;
    int diff = p1 - p0;
    float tz0, tz1, tz2;
    if (diff == 1 || diff == -2) {           // cross(tx, y)
        tz0 = tx1*y2 - tx2*y1; tz1 = tx2*y0 - tx0*y2; tz2 = tx0*y1 - tx1*y0;
    } else {                                 // cross(y, tx)
        tz0 = y1*tx2 - y2*tx1; tz1 = y2*tx0 - y0*tx2; tz2 = y0*tx1 - y1*tx0;
    }
    float itzn = rsqrtf(tz0*tz0 + tz1*tz1 + tz2*tz2);
    tz0 *= itzn; tz1 *= itzn; tz2 *= itzn;

    if (jj == 0) {
        out[n*9 + 0 + p0] = tx0; out[n*9 + 3 + p0] = tx1; out[n*9 + 6 + p0] = tx2;
        out[n*9 + 0 + p1] = y0;  out[n*9 + 3 + p1] = y1;  out[n*9 + 6 + p1] = y2;
        out[n*9 + 0 + p2] = tz0; out[n*9 + 3 + p2] = tz1; out[n*9 + 6 + p2] = tz2;
        float* tvec = out + (size_t)9 * N;
        tvec[n*6+0] = vec[n*6+0];
        tvec[n*6+1] = vec[n*6+1];
        tvec[n*6+2] = vec[n*6+2];
        // real_s = exp(rs)*0 + 1 == 1 exactly, so scale is identity, ldj += 0
        tvec[n*6+3] = vec[n*6+3] + sConds[s][259];
        tvec[n*6+4] = vec[n*6+4] + sConds[s][260];
        tvec[n*6+5] = vec[n*6+5] + sConds[s][261];
        out[(size_t)15 * N + n] = ldj;
    }
}

extern "C" void kernel_launch(void* const* d_in, const int* in_sizes, int n_in,
                              void* d_out, int out_size, void* d_ws, size_t ws_size,
                              hipStream_t stream)
{
    (void)n_in; (void)out_size; (void)ws_size;
    const float* rot  = (const float*)d_in[0];
    const float* vecp = (const float*)d_in[1];
    const float* feat = (const float*)d_in[2];
    const int*   perm = (const int*)  d_in[3];
    const float* W1   = (const float*)d_in[4];
    const float* b1   = (const float*)d_in[5];
    const float* W2   = (const float*)d_in[6];
    const float* b2   = (const float*)d_in[7];

    const int N = in_sizes[0] / 9;                 // rotation is (N,3,3)

    _Float16* w1f = (_Float16*)d_ws;               // 147456 B
    _Float16* w2f = w1f + W1F_ELEMS;               // 139264 B (total < 287 KB)

    const int prep_threads = W1F_ELEMS + W2F_ELEMS;   // 143360
    prep_weights<<<(prep_threads + 255) / 256, 256, 0, stream>>>(W1, W2, w1f, w2f);

    mobius_fused<<<N / 16, 256, 0, stream>>>(rot, vecp, feat, perm, b1, b2,
                                             w1f, w2f, (float*)d_out, N);
}
